// AdvancedSSMapper_38414187495430
// MI455X (gfx1250) — matrix operand order repaired and checked by reading, not compiled, not run on hardware
//
#include <hip/hip_runtime.h>

#define B_  8
#define S_  1024
#define F_  1280
#define H_  256
#define E_  512
#define NH_ 8
#define BS_ (B_ * S_)

typedef _Float16 h16;
typedef __attribute__((ext_vector_type(16))) _Float16 v16h;
typedef __attribute__((ext_vector_type(8)))  _Float16 v8h;
typedef __attribute__((ext_vector_type(8)))  float    v8f;

__device__ __forceinline__ v8f wmma_f32_f16(v16h a, v16h b, v8f c) {
    // D = A(16x32 f16) * B(32x16 f16) + C(16x16 f32)
    return __builtin_amdgcn_wmma_f32_16x16x32_f16(false, a, false, b, (short)0, c, false, false);
}

__device__ __forceinline__ float sigmoidf_(float x) { return 1.0f / (1.0f + __expf(-x)); }

// ---------------------------------------------------------------------------
// Generic WMMA GEMM: C[M,N] = act(A[M,K] @ W[N,K]^T + bias), M=BS_, tiles 64x64x32
// ---------------------------------------------------------------------------
#define GF_BIAS 1
#define GF_RELU 2
#define GF_ACC  4

__global__ __launch_bounds__(256)
void k_wmma_gemm(const h16* __restrict__ A, int lda,
                 const h16* __restrict__ W, int ldw,
                 const float* __restrict__ bias,
                 float* __restrict__ C32, h16* __restrict__ C16,
                 int ldc, int K, int flags)
{
    __shared__ h16 sA[64 * 32];
    __shared__ h16 sB[64 * 32];

    const int tid  = threadIdx.x;
    const int wave = tid >> 5;
    const int lane = tid & 31;
    const int bm = blockIdx.x * 64;
    const int bn = blockIdx.y * 64;
    const int wm = (wave >> 1) * 16;   // 0,16,32,48
    const int wn = (wave & 1) * 32;    // 0,32
    const int srow = tid >> 2;         // 0..63
    const int scol = (tid & 3) * 8;    // 0,8,16,24
    const int r  = lane & 15;
    const int a0 = (lane < 16) ? 0 : 16;  // f16 A-frag K base (contiguous layout)
    const int b0 = (lane < 16) ? 0 : 16;  // f16 B-frag K base (contiguous layout)

    v8f acc0 = {};
    v8f acc1 = {};

    for (int kt = 0; kt < K; kt += 32) {
        *(v8h*)&sA[srow * 32 + scol] =
            *(const v8h*)&A[(long long)(bm + srow) * lda + kt + scol];
        *(v8h*)&sB[srow * 32 + scol] =
            *(const v8h*)&W[(long long)(bn + srow) * ldw + kt + scol];
        if (kt + 32 < K)
            __builtin_prefetch(&A[(long long)(bm + srow) * lda + kt + 32 + scol], 0, 1);
        __syncthreads();

        v16h fa, fb0, fb1;
        ((v8h*)&fa)[0]  = *(const v8h*)&sA[(wm + r) * 32 + a0];
        ((v8h*)&fa)[1]  = *(const v8h*)&sA[(wm + r) * 32 + a0 + 8];
        ((v8h*)&fb0)[0] = *(const v8h*)&sB[(wn + r) * 32 + b0];
        ((v8h*)&fb0)[1] = *(const v8h*)&sB[(wn + r) * 32 + b0 + 8];
        ((v8h*)&fb1)[0] = *(const v8h*)&sB[(wn + 16 + r) * 32 + b0];
        ((v8h*)&fb1)[1] = *(const v8h*)&sB[(wn + 16 + r) * 32 + b0 + 8];

        acc0 = wmma_f32_f16(fa, fb0, acc0);
        acc1 = wmma_f32_f16(fa, fb1, acc1);
        __syncthreads();
    }

    const int rofs = (lane < 16) ? 0 : 8;
    for (int half = 0; half < 2; ++half) {
        v8f acc;
        if (half) acc = acc1; else acc = acc0;
        const int ncol = bn + wn + half * 16 + r;
        const float bv = (flags & GF_BIAS) ? bias[ncol] : 0.0f;
        #pragma unroll
        for (int j = 0; j < 8; ++j) {
            const long long ci = (long long)(bm + wm + rofs + j) * ldc + ncol;
            float v = acc[j] + bv;
            if (flags & GF_ACC)  v += C32[ci];
            if (flags & GF_RELU) v = fmaxf(v, 0.0f);
            if (C32) C32[ci] = v;
            if (C16) C16[ci] = (h16)v;
        }
    }
}

// ---------------------------------------------------------------------------
// Persistent LSTM scan (one workgroup per direction). Gates via WMMA:
// g[16x1024] = h[16x256] @ Whh[1024x256]^T  (batch 8 padded to 16 rows)
// ---------------------------------------------------------------------------
__global__ __launch_bounds__(256)
void k_lstm_scan(const float* __restrict__ xg,    // [B_,S_,1024]  precomputed x@Wih^T + b
                 const h16* __restrict__ whh,     // [1024,256] f16
                 h16* __restrict__ y16, int ld16, // pre-offset to direction column base
                 float* __restrict__ y32, int ld32,
                 int reverse)
{
    __shared__ h16   sh[16 * 256];   // h state (rows 8..15 stay zero)
    __shared__ float sc[8 * 256];    // c state
    __shared__ float sg[8 * 1024];   // gates

    const int tid  = threadIdx.x;
    const int wave = tid >> 5;
    const int lane = tid & 31;
    const int r  = lane & 15;
    const int a0 = (lane < 16) ? 0 : 16;
    const int b0 = (lane < 16) ? 0 : 16;

    for (int i = tid; i < 16 * 256; i += 256) sh[i] = (h16)0.0f;
    for (int i = tid; i < 8 * 256;  i += 256) sc[i] = 0.0f;
    __syncthreads();

    for (int t = 0; t < S_; ++t) {
        const int ts = reverse ? (S_ - 1 - t) : t;
        for (int nt = 0; nt < 8; ++nt) {
            const int ncol = wave * 128 + nt * 16;
            v8f acc = {};
            #pragma unroll
            for (int kt = 0; kt < 256; kt += 32) {
                v16h fa, fb;
                ((v8h*)&fa)[0] = *(const v8h*)&sh[r * 256 + kt + a0];
                ((v8h*)&fa)[1] = *(const v8h*)&sh[r * 256 + kt + a0 + 8];
                ((v8h*)&fb)[0] = *(const v8h*)&whh[(long long)(ncol + r) * 256 + kt + b0];
                ((v8h*)&fb)[1] = *(const v8h*)&whh[(long long)(ncol + r) * 256 + kt + b0 + 8];
                acc = wmma_f32_f16(fa, fb, acc);
            }
            if (lane < 16) {   // rows 0..7 are the real batch
                #pragma unroll
                for (int j = 0; j < 8; ++j)
                    sg[j * 1024 + ncol + r] =
                        acc[j] + xg[((long long)j * S_ + ts) * 1024 + ncol + r];
            }
        }
        __syncthreads();
        for (int u = tid; u < 8 * 256; u += 256) {
            const int bb = u >> 8, hu = u & 255;
            const float gi = sg[bb * 1024 + hu];
            const float gf = sg[bb * 1024 + 256 + hu];
            const float gg = sg[bb * 1024 + 512 + hu];
            const float go = sg[bb * 1024 + 768 + hu];
            float c = sc[u];
            c = sigmoidf_(gf) * c + sigmoidf_(gi) * tanhf(gg);
            const float h = sigmoidf_(go) * tanhf(c);
            sc[u] = c;
            sh[bb * 256 + hu] = (h16)h;
            const long long o = (long long)bb * S_ + ts;
            y16[o * ld16 + hu] = (h16)h;
            if (y32) y32[o * ld32 + hu] = h;
        }
        __syncthreads();
    }
}

// ---------------------------------------------------------------------------
// Attention: one block per (batch, head, 8 query rows). WMMA QK^T + PV.
// ---------------------------------------------------------------------------
__global__ __launch_bounds__(256)
void k_attention(const h16* __restrict__ qkv,  // [BS_,1536] f16 (q|k|v)
                 h16* __restrict__ ctx)        // [BS_,512]  f16
{
    __shared__ float sS[8 * 1024];   // scores -> probs
    __shared__ h16   sQ[16 * 64];
    __shared__ h16   sV[32 * 64];
    __shared__ float sred[8 * 32];

    const int tid  = threadIdx.x;
    const int wave = tid >> 5;
    const int lane = tid & 31;
    const int r  = lane & 15;
    const int a0 = (lane < 16) ? 0 : 16;
    const int b0 = (lane < 16) ? 0 : 16;

    const int bb   = blockIdx.z;
    const int head = blockIdx.y;
    const int q0   = blockIdx.x * 8;

    const h16* qb = qkv + (long long)bb * S_ * 1536 + head * 64;
    const h16* kb = qb + 512;
    const h16* vb = qb + 1024;

    if (tid < 128) {                 // stage Q tile (rows 8..15 zero pad)
        const int qr = tid >> 3, qc = (tid & 7) * 8;
        v8h val = {};
        if (qr < 8) val = *(const v8h*)&qb[(long long)(q0 + qr) * 1536 + qc];
        *(v8h*)&sQ[qr * 64 + qc] = val;
    }
    __syncthreads();

    // scores = Q K^T / 8  : each wave covers 128 key columns
    for (int nt = 0; nt < 8; ++nt) {
        const int n0 = wave * 128 + nt * 16;
        v8f acc = {};
        #pragma unroll
        for (int kt = 0; kt < 64; kt += 32) {
            v16h fa, fb;
            ((v8h*)&fa)[0] = *(const v8h*)&sQ[r * 64 + kt + a0];
            ((v8h*)&fa)[1] = *(const v8h*)&sQ[r * 64 + kt + a0 + 8];
            ((v8h*)&fb)[0] = *(const v8h*)&kb[(long long)(n0 + r) * 1536 + kt + b0];
            ((v8h*)&fb)[1] = *(const v8h*)&kb[(long long)(n0 + r) * 1536 + kt + b0 + 8];
            acc = wmma_f32_f16(fa, fb, acc);
        }
        if (lane < 16) {
            #pragma unroll
            for (int j = 0; j < 8; ++j)
                sS[j * 1024 + n0 + r] = acc[j] * 0.125f;
        }
    }
    __syncthreads();

    // softmax over 1024 keys: 8 rows x 32 threads x 32 cols
    {
        const int row = tid >> 5;
        const int seg = (tid & 31) * 32;
        float mx = -3.0e38f;
        for (int i = 0; i < 32; ++i) mx = fmaxf(mx, sS[row * 1024 + seg + i]);
        sred[row * 32 + (tid & 31)] = mx;
        __syncthreads();
        float rmax = -3.0e38f;
        for (int i = 0; i < 32; ++i) rmax = fmaxf(rmax, sred[row * 32 + i]);
        __syncthreads();
        float sum = 0.0f;
        for (int i = 0; i < 32; ++i) {
            const float e = __expf(sS[row * 1024 + seg + i] - rmax);
            sS[row * 1024 + seg + i] = e;
            sum += e;
        }
        sred[row * 32 + (tid & 31)] = sum;
        __syncthreads();
        float rsum = 0.0f;
        for (int i = 0; i < 32; ++i) rsum += sred[row * 32 + i];
        const float inv = 1.0f / rsum;
        for (int i = 0; i < 32; ++i) sS[row * 1024 + seg + i] *= inv;
    }
    __syncthreads();

    // ctx = P V : waves 0..3 each own a 16-wide d-tile, all waves stage V
    v8f oacc = {};
    for (int kt = 0; kt < 1024; kt += 32) {
        {
            const int vr = tid >> 3, vc = (tid & 7) * 8;
            *(v8h*)&sV[vr * 64 + vc] = *(const v8h*)&vb[(long long)(kt + vr) * 1536 + vc];
        }
        __syncthreads();
        if (wave < 4) {
            v16h fa, fb;
            #pragma unroll
            for (int i = 0; i < 8; ++i) {
                const float p0 = (r < 8) ? sS[r * 1024 + kt + a0 + i]      : 0.0f;
                const float p1 = (r < 8) ? sS[r * 1024 + kt + a0 + 8 + i] : 0.0f;
                ((h16*)&fa)[i]     = (h16)p0;
                ((h16*)&fa)[i + 8] = (h16)p1;
            }
            const int d = wave * 16 + r;
            #pragma unroll
            for (int i = 0; i < 16; ++i)
                ((h16*)&fb)[i] = sV[(b0 + i) * 64 + d];
            oacc = wmma_f32_f16(fa, fb, oacc);
        }
        __syncthreads();
    }
    if (wave < 4 && lane < 16) {
        const int d = head * 64 + wave * 16 + r;
        #pragma unroll
        for (int j = 0; j < 8; ++j)
            ctx[(long long)(bb * S_ + q0 + j) * 512 + d] = (h16)oacc[j];
    }
}

// ---------------------------------------------------------------------------
// Small utility / feature kernels (scalar VALU; negligible FLOPs)
// ---------------------------------------------------------------------------
__global__ void k_cvt_f32f16(const float* __restrict__ s, h16* __restrict__ d, long long n) {
    const long long i = (long long)blockIdx.x * blockDim.x + threadIdx.x;
    if (i < n) d[i] = (h16)s[i];
}

__global__ void k_conv_repack(const float* __restrict__ w, h16* __restrict__ d, int k) {
    const long long i = (long long)blockIdx.x * blockDim.x + threadIdx.x;
    const long long total = (long long)H_ * F_ * k;
    if (i < total) {
        const int t = (int)(i % k);
        const long long hf = i / k;
        d[(long long)t * H_ * F_ + hf] = (h16)w[i];   // [H,F,k] -> [k][H][F]
    }
}

__global__ void k_bias_sum(const float* __restrict__ a, const float* __restrict__ b,
                           float* __restrict__ o, int n) {
    const int i = blockIdx.x * blockDim.x + threadIdx.x;
    if (i < n) o[i] = a[i] + b[i];
}

__global__ void k_add_combined(const float* __restrict__ a, const float* __restrict__ b,
                               float* __restrict__ o32, h16* __restrict__ fus) {
    const long long i = (long long)blockIdx.x * blockDim.x + threadIdx.x;
    if (i < (long long)BS_ * E_) {
        const float v = a[i] + b[i];
        o32[i] = v;
        const long long row = i >> 9, col = i & 511;
        fus[row * 640 + col] = (h16)v;
    }
}

__global__ void k_add_f16out(const float* __restrict__ a, const float* __restrict__ b,
                             h16* __restrict__ o16) {
    const long long i = (long long)blockIdx.x * blockDim.x + threadIdx.x;
    if (i < (long long)BS_ * E_) o16[i] = (h16)(a[i] + b[i]);
}

__global__ void k_dist_feat(const float* __restrict__ coords,
                            const float* __restrict__ w1, const float* __restrict__ b1,
                            const float* __restrict__ w2, const float* __restrict__ b2,
                            h16* __restrict__ fus) {
    const int idx = blockIdx.x * blockDim.x + threadIdx.x;
    if (idx >= BS_) return;
    const int bb = idx >> 10, s = idx & 1023;
    const float* cb = coords + (long long)bb * S_ * 3;
    const float x0 = cb[s * 3 + 0], y0 = cb[s * 3 + 1], z0 = cb[s * 3 + 2];
    float acc[64];
    for (int o = 0; o < 64; ++o) acc[o] = 0.0f;
    for (int off = 1; off <= 4; ++off) {
        float dd = 0.0f;                 // padded region: d = 0, MLP still applied
        if (s + off < S_) {
            const float dx = cb[(s + off) * 3 + 0] - x0;
            const float dy = cb[(s + off) * 3 + 1] - y0;
            const float dz = cb[(s + off) * 3 + 2] - z0;
            dd = sqrtf(dx * dx + dy * dy + dz * dz);
        }
        float h1[32];
        for (int j = 0; j < 32; ++j) h1[j] = fmaxf(w1[j] * dd + b1[j], 0.0f);
        for (int o = 0; o < 64; ++o) {
            float v = b2[o];
            for (int j = 0; j < 32; ++j) v += w2[o * 32 + j] * h1[j];
            acc[o] += v;
        }
    }
    for (int o = 0; o < 64; ++o)
        fus[(long long)idx * 640 + 512 + o] = (h16)(acc[o] * 0.25f);
}

__device__ __forceinline__ float angle_between(float ax, float ay, float az,
                                               float bx, float by, float bz) {
    const float num = ax * bx + ay * by + az * bz;
    const float na = fmaxf(sqrtf(ax * ax + ay * ay + az * az), 1e-8f);
    const float nb = fmaxf(sqrtf(bx * bx + by * by + bz * bz), 1e-8f);
    float c = num / (na * nb);
    c = fminf(fmaxf(c, -1.0f), 1.0f);
    return acosf(c);
}

__global__ void k_ang_feat(const float* __restrict__ coords,
                           const float* __restrict__ w1, const float* __restrict__ b1,
                           const float* __restrict__ w2, const float* __restrict__ b2,
                           h16* __restrict__ fus) {
    const int idx = blockIdx.x * blockDim.x + threadIdx.x;
    if (idx >= BS_) return;
    const int bb = idx >> 10, s = idx & 1023;
    const float* cb = coords + (long long)bb * S_ * 3;
    float an0 = 0.0f, an1 = 0.0f;
    if (s >= 1 && s <= S_ - 2) {
        float u[3], v[3];
        for (int d = 0; d < 3; ++d) {
            u[d] = cb[s * 3 + d]       - cb[(s - 1) * 3 + d];
            v[d] = cb[(s + 1) * 3 + d] - cb[s * 3 + d];
        }
        an0 = angle_between(u[0], u[1], u[2], v[0], v[1], v[2]);
        if (s <= S_ - 3) {
            float w[3];
            for (int d = 0; d < 3; ++d) w[d] = cb[(s + 2) * 3 + d] - cb[(s + 1) * 3 + d];
            const float n0x = u[1]*v[2]-u[2]*v[1], n0y = u[2]*v[0]-u[0]*v[2], n0z = u[0]*v[1]-u[1]*v[0];
            const float n1x = v[1]*w[2]-v[2]*w[1], n1y = v[2]*w[0]-v[0]*w[2], n1z = v[0]*w[1]-v[1]*w[0];
            an1 = angle_between(n0x, n0y, n0z, n1x, n1y, n1z);
        }
    }
    float h1[32];
    for (int j = 0; j < 32; ++j)
        h1[j] = fmaxf(w1[j * 2] * an0 + w1[j * 2 + 1] * an1 + b1[j], 0.0f);
    for (int o = 0; o < 64; ++o) {
        float v = b2[o];
        for (int j = 0; j < 32; ++j) v += w2[o * 32 + j] * h1[j];
        fus[(long long)idx * 640 + 576 + o] = (h16)v;
    }
}

__global__ void k_cls_out(const h16* __restrict__ h, const float* __restrict__ w2,
                          const float* __restrict__ b2, float* __restrict__ out) {
    const int idx = blockIdx.x * blockDim.x + threadIdx.x;
    if (idx >= BS_) return;
    const h16* hr = h + (long long)idx * H_;
    for (int c = 0; c < 8; ++c) {
        float v = b2[c];
        for (int u = 0; u < H_; ++u) v += w2[c * H_ + u] * (float)hr[u];
        out[(long long)idx * 8 + c] = v;
    }
}

// ---------------------------------------------------------------------------
// Host orchestration
// ---------------------------------------------------------------------------
extern "C" void kernel_launch(void* const* d_in, const int* in_sizes, int n_in,
                              void* d_out, int out_size, void* d_ws, size_t ws_size,
                              hipStream_t stream)
{
    (void)in_sizes; (void)n_in; (void)out_size; (void)ws_size;
    const float* features = (const float*)d_in[0];
    const float* coords   = (const float*)d_in[1];
    const int kk[3] = {3, 7, 15};
    const float* conv_w[3]; const float* conv_b[3];
    for (int i = 0; i < 3; ++i) {
        conv_w[i] = (const float*)d_in[2 + 2 * i];
        conv_b[i] = (const float*)d_in[3 + 2 * i];
    }
    const float *wih[6], *whh[6], *bih[6], *bhh[6];
    for (int li = 0; li < 6; ++li) {
        const int base = 8 + li * 4;
        wih[li] = (const float*)d_in[base + 0];
        whh[li] = (const float*)d_in[base + 1];
        bih[li] = (const float*)d_in[base + 2];
        bhh[li] = (const float*)d_in[base + 3];
    }
    const float* attn_in_w  = (const float*)d_in[32];
    const float* attn_in_b  = (const float*)d_in[33];
    const float* attn_out_w = (const float*)d_in[34];
    const float* attn_out_b = (const float*)d_in[35];
    const float* dist_w1 = (const float*)d_in[36];
    const float* dist_b1 = (const float*)d_in[37];
    const float* dist_w2 = (const float*)d_in[38];
    const float* dist_b2 = (const float*)d_in[39];
    const float* ang_w1  = (const float*)d_in[40];
    const float* ang_b1  = (const float*)d_in[41];
    const float* ang_w2  = (const float*)d_in[42];
    const float* ang_b2  = (const float*)d_in[43];
    const float* fus_w1  = (const float*)d_in[44];
    const float* fus_b1  = (const float*)d_in[45];
    const float* fus_w2  = (const float*)d_in[46];
    const float* fus_b2  = (const float*)d_in[47];
    const float* cls_w1  = (const float*)d_in[48];
    const float* cls_b1  = (const float*)d_in[49];
    const float* cls_w2  = (const float*)d_in[50];
    const float* cls_b2  = (const float*)d_in[51];

    char* wsp = (char*)d_ws;
    size_t cur = 0;
    auto alloc = [&](size_t bytes) -> void* {
        cur = (cur + 255) & ~(size_t)255;
        void* p = (void*)(wsp + cur);
        cur += bytes;
        return p;
    };

    const long long GUARD = 16LL * F_;   // guard rows for shifted conv-tap GEMMs
    h16* feat16g = (h16*)alloc(((long long)BS_ * F_ + 2 * GUARD) * sizeof(h16));
    h16* feat16  = feat16g + GUARD;
    float* convout32  = (float*)alloc((long long)BS_ * 768 * sizeof(float));
    h16*   convout16  = (h16*)  alloc((long long)BS_ * 768 * sizeof(h16));
    float* xgf        = (float*)alloc((long long)BS_ * 1024 * sizeof(float));
    float* xgb        = (float*)alloc((long long)BS_ * 1024 * sizeof(float));
    h16*   x16a       = (h16*)  alloc((long long)BS_ * 512 * sizeof(h16));
    h16*   x16b       = (h16*)  alloc((long long)BS_ * 512 * sizeof(h16));
    float* lstm_out32 = (float*)alloc((long long)BS_ * 512 * sizeof(float));
    h16*   qkv16      = (h16*)  alloc((long long)BS_ * 1536 * sizeof(h16));
    h16*   ctx16      = (h16*)  alloc((long long)BS_ * 512 * sizeof(h16));
    float* attn_out32 = (float*)alloc((long long)BS_ * 512 * sizeof(float));
    float* combined32 = (float*)alloc((long long)BS_ * 512 * sizeof(float));
    h16*   fus_in16   = (h16*)  alloc((long long)BS_ * 640 * sizeof(h16));
    h16*   h1_16      = (h16*)  alloc((long long)BS_ * 512 * sizeof(h16));
    float* sf32       = (float*)alloc((long long)BS_ * 512 * sizeof(float));
    h16*   comb2_16   = (h16*)  alloc((long long)BS_ * 512 * sizeof(h16));
    h16*   clsh16     = (h16*)  alloc((long long)BS_ * 256 * sizeof(h16));
    h16* convw16[3];
    for (int i = 0; i < 3; ++i)
        convw16[i] = (h16*)alloc((long long)H_ * F_ * kk[i] * sizeof(h16));
    h16* wih16[6]; h16* whh16[6]; float* bsum[6];
    for (int li = 0; li < 6; ++li) {
        const int din = (li < 2) ? 768 : 512;
        wih16[li] = (h16*)alloc((long long)1024 * din * sizeof(h16));
        whh16[li] = (h16*)alloc((long long)1024 * 256 * sizeof(h16));
        bsum[li]  = (float*)alloc(1024 * sizeof(float));
    }
    h16* attn_in_w16  = (h16*)alloc(1536LL * 512 * sizeof(h16));
    h16* attn_out_w16 = (h16*)alloc(512LL * 512 * sizeof(h16));
    h16* fus_w1_16    = (h16*)alloc(512LL * 640 * sizeof(h16));
    h16* fus_w2_16    = (h16*)alloc(512LL * 512 * sizeof(h16));
    h16* cls_w1_16    = (h16*)alloc(256LL * 512 * sizeof(h16));

    auto cvt = [&](const float* s, h16* dptr, long long n) {
        k_cvt_f32f16<<<(int)((n + 255) / 256), 256, 0, stream>>>(s, dptr, n);
    };

    // precision staging: f32 -> f16 for all WMMA operands
    cvt(features, feat16, (long long)BS_ * F_);
    for (int i = 0; i < 3; ++i) {
        const long long n = (long long)H_ * F_ * kk[i];
        k_conv_repack<<<(int)((n + 255) / 256), 256, 0, stream>>>(conv_w[i], convw16[i], kk[i]);
    }
    for (int li = 0; li < 6; ++li) {
        const int din = (li < 2) ? 768 : 512;
        cvt(wih[li], wih16[li], (long long)1024 * din);
        cvt(whh[li], whh16[li], 1024LL * 256);
        k_bias_sum<<<4, 256, 0, stream>>>(bih[li], bhh[li], bsum[li], 1024);
    }
    cvt(attn_in_w,  attn_in_w16,  1536LL * 512);
    cvt(attn_out_w, attn_out_w16, 512LL * 512);
    cvt(fus_w1, fus_w1_16, 512LL * 640);
    cvt(fus_w2, fus_w2_16, 512LL * 512);
    cvt(cls_w1, cls_w1_16, 256LL * 512);

    auto gemm = [&](const h16* A, int lda, const h16* W, int ldw, const float* bias,
                    float* C32, h16* C16, int ldc, int N, int K, int flags) {
        dim3 grid(BS_ / 64, N / 64);
        k_wmma_gemm<<<grid, 256, 0, stream>>>(A, lda, W, ldw, bias, C32, C16, ldc, K, flags);
    };

    // multi-kernel conv bank = 25 shifted-A tap GEMMs into [BS,768]
    for (int i = 0; i < 3; ++i) {
        const int k = kk[i], pad = k / 2;
        for (int t = 0; t < k; ++t) {
            int flags = (t == 0) ? GF_BIAS : GF_ACC;
            h16* c16 = nullptr;
            if (t == k - 1) { flags |= GF_RELU; c16 = convout16 + i * 256; }
            gemm(feat16 + (long long)(t - pad) * F_, F_,
                 convw16[i] + (long long)t * H_ * F_, F_,
                 conv_b[i],
                 convout32 + i * 256, c16, 768, 256, F_, flags);
        }
    }

    // 3-layer biLSTM: bulk input GEMM + persistent WMMA scan per direction
    const h16* xin = convout16;
    int kdim = 768;
    h16* xcat[3] = {x16a, x16b, x16a};
    for (int l = 0; l < 3; ++l) {
        const int lf = l * 2, lb = l * 2 + 1;
        gemm(xin, kdim, wih16[lf], kdim, bsum[lf], xgf, nullptr, 1024, 1024, kdim, GF_BIAS);
        gemm(xin, kdim, wih16[lb], kdim, bsum[lb], xgb, nullptr, 1024, 1024, kdim, GF_BIAS);
        float* y32f = (l == 2) ? lstm_out32       : nullptr;
        float* y32b = (l == 2) ? lstm_out32 + 256 : nullptr;
        k_lstm_scan<<<1, 256, 0, stream>>>(xgf, whh16[lf], xcat[l],       512, y32f, 512, 0);
        k_lstm_scan<<<1, 256, 0, stream>>>(xgb, whh16[lb], xcat[l] + 256, 512, y32b, 512, 1);
        xin = xcat[l]; kdim = 512;
    }
    // xin == x16a == lstm_out (f16); lstm_out32 holds f32 copy

    // attention
    gemm(xin, 512, attn_in_w16, 512, attn_in_b, nullptr, qkv16, 1536, 1536, 512, GF_BIAS);
    k_attention<<<dim3(S_ / 8, NH_, B_), 256, 0, stream>>>(qkv16, ctx16);
    gemm(ctx16, 512, attn_out_w16, 512, attn_out_b, attn_out32, nullptr, 512, 512, 512, GF_BIAS);

    {
        const long long n = (long long)BS_ * 512;
        k_add_combined<<<(int)((n + 255) / 256), 256, 0, stream>>>(lstm_out32, attn_out32,
                                                                   combined32, fus_in16);
    }
    k_dist_feat<<<BS_ / 256, 256, 0, stream>>>(coords, dist_w1, dist_b1, dist_w2, dist_b2, fus_in16);
    k_ang_feat <<<BS_ / 256, 256, 0, stream>>>(coords, ang_w1, ang_b1, ang_w2, ang_b2, fus_in16);

    // fusion MLP + residual + classifier
    gemm(fus_in16, 640, fus_w1_16, 640, fus_b1, nullptr, h1_16, 512, 512, 640, GF_BIAS | GF_RELU);
    gemm(h1_16, 512, fus_w2_16, 512, fus_b2, sf32, nullptr, 512, 512, 512, GF_BIAS);
    {
        const long long n = (long long)BS_ * 512;
        k_add_f16out<<<(int)((n + 255) / 256), 256, 0, stream>>>(combined32, sf32, comb2_16);
    }
    gemm(comb2_16, 512, cls_w1_16, 512, cls_b1, nullptr, clsh16, 256, 256, 512, GF_BIAS | GF_RELU);
    k_cls_out<<<BS_ / 256, 256, 0, stream>>>(clsh16, cls_w2, cls_b2, (float*)d_out);
}
